// HeterogeneousCompGCN_88261577933338
// MI455X (gfx1250) — compile-verified, matrix operand order checked by reading, not run on hardware
//
#include <hip/hip_runtime.h>

#define N_NODES  100000
#define N_TYPE_A 50000
#define N_TYPE_B 50000
#define N_EDGES  600000
#define D_IN_A   256
#define D        128
#define N_REL2   32
#define LN_EPS   1e-5f

typedef __attribute__((ext_vector_type(16))) _Float16 v16h;
typedef __attribute__((ext_vector_type(8)))  _Float16 v8h;
typedef __attribute__((ext_vector_type(4)))  _Float16 v4h;
typedef __attribute__((ext_vector_type(8)))  float    v8f;

#define CDIV(a, b) (((a) + (b) - 1) / (b))
#define SCATTER_BLOCKS 2048

// async global->LDS 16B copy, per-lane addresses; tracked by ASYNCcnt
__device__ __forceinline__ void async_g2l_b128(unsigned lds_byte_addr, const void* gaddr) {
  asm volatile("global_load_async_to_lds_b128 %0, %1, off"
               :: "v"(lds_byte_addr), "v"(gaddr) : "memory");
}
__device__ __forceinline__ void async_wait0() {
  asm volatile("s_wait_asynccnt 0x0" ::: "memory");
}

// ---------------------------------------------------------------- utilities
__global__ void k_zero_f32(float* __restrict__ p, int n) {
  int i = blockIdx.x * blockDim.x + threadIdx.x;
  if (i < n) p[i] = 0.0f;
}

__global__ void k_f32_to_f16(const float* __restrict__ s, _Float16* __restrict__ d, int n) {
  int i = blockIdx.x * blockDim.x + threadIdx.x;
  if (i < n) d[i] = (_Float16)s[i];
}

// degree counting: deg_s[src[e]] += 1, deg_t[dst[e]] += 1 (into zeroed bufs)
__global__ void k_degree(const int* __restrict__ src, const int* __restrict__ dst,
                         float* deg_s, float* deg_t, int E) {
  int i = blockIdx.x * blockDim.x + threadIdx.x;
  if (i < E) {
    atomicAdd(&deg_s[src[i]], 1.0f);
    atomicAdd(&deg_t[dst[i]], 1.0f);
  }
}

__global__ void k_invsqrt(float* __restrict__ p, int n) {
  int i = blockIdx.x * blockDim.x + threadIdx.x;
  if (i < n) {
    float d = p[i];
    p[i] = (d > 0.0f) ? rsqrtf(d) : 0.0f;
  }
}

// tiny f32 GEMM C[M,N] = A[M,K] @ W[N,K]^T (rel/loop tables, full precision)
__global__ void k_small_gemm(const float* __restrict__ A, const float* __restrict__ W,
                             float* __restrict__ C, int M, int N, int K) {
  int i = blockIdx.x * blockDim.x + threadIdx.x;
  if (i >= M * N) return;
  int m = i / N, n = i - m * N;
  const float* a = A + (size_t)m * K;
  const float* w = W + (size_t)n * K;
  float s = 0.0f;
  for (int k = 0; k < K; ++k) s += a[k] * w[k];
  C[i] = s;
}

// ---------------------------------------------------- WMMA GEMM  Y = X @ Wh^T (+bias)
// Xh: [M,K] f16 row-major, Wh: [128,K] f16 row-major.  Output either f32 (Yf) or f16 (Yh).
// Block = 256 threads = 8 waves; block -> 16 rows (async-staged in LDS); wave w -> cols [16w,16w+16).
__global__ __launch_bounds__(256)
void k_gemm_wmma(const _Float16* __restrict__ Xh, const _Float16* __restrict__ Wh,
                 const float* __restrict__ bias, float* __restrict__ Yf,
                 _Float16* __restrict__ Yh, int K) {
  __shared__ __align__(16) _Float16 Ah[16 * (256 + 8)];
  const int ldA  = K + 8;                 // halves; multiple of 8 keeps 16B runs aligned
  const int row0 = blockIdx.x * 16;
  const int tid  = threadIdx.x;

  // stage 16 x K f16 tile via async global->LDS (16B per transfer)
  const unsigned lbase  = (unsigned)(uintptr_t)&Ah[0];
  const int      chunks = 2 * K;          // 16*K/8 16B-chunks
  const int      cpr    = K / 8;          // chunks per row
  for (int i = tid; i < chunks; i += 256) {
    int r = i / cpr, c = (i - r * cpr) * 8;
    async_g2l_b128(lbase + (unsigned)((r * ldA + c) * 2),
                   Xh + (size_t)(row0 + r) * K + c);
  }
  async_wait0();
  __syncthreads();

  const int wv   = tid >> 5;
  const int lane = tid & 31;
  const int g    = lane >> 4;             // lane group selects K-half
  const int mr   = lane & 15;             // A row / B column within tile
  const int col0 = wv * 16;

  v8f acc = {};
  for (int kb = 0; kb < K; kb += 32) {
    // ISA 16-bit layout: lane-group g holds K = kb + [8g,8g+8) and kb+16+[8g,8g+8)
    const _Float16* ar = &Ah[mr * ldA + kb];
    const _Float16* br = &Wh[(size_t)(col0 + mr) * K + kb];
    v8h a0 = *(const v8h*)(ar + 8 * g);
    v8h a1 = *(const v8h*)(ar + 16 + 8 * g);
    v8h b0 = *(const v8h*)(br + 8 * g);
    v8h b1 = *(const v8h*)(br + 16 + 8 * g);
    v16h a = __builtin_shufflevector(a0, a1, 0,1,2,3,4,5,6,7,8,9,10,11,12,13,14,15);
    v16h b = __builtin_shufflevector(b0, b1, 0,1,2,3,4,5,6,7,8,9,10,11,12,13,14,15);
    acc = __builtin_amdgcn_wmma_f32_16x16x32_f16(false, a, false, b,
                                                 (short)0, acc, false, false);
  }

  float bv = bias ? bias[col0 + mr] : 0.0f;
#pragma unroll
  for (int j = 0; j < 8; ++j) {           // D VGPR j -> row j+8g, col = lane&15
    size_t idx = (size_t)(row0 + j + 8 * g) * D + col0 + mr;
    float  val = acc[j] + bv;
    if (Yf) Yf[idx] = val;
    else    Yh[idx] = (_Float16)val;
  }
}

// ---------------------------------------------------------------- edge scatter
// Fixed grid, grid-stride over edges; RelT (16 KB) async-staged in LDS once per block.
// 1 wave = 1 edge; lane = one float4 chunk.  acc[to] += (Y[from]-RelT[r]) * inv_f[from]*inv_t[to]
__global__ __launch_bounds__(256)
void k_scatter(const float4* __restrict__ Y, const float* __restrict__ RelT,
               const int* __restrict__ efrom, const int* __restrict__ eto,
               const int* __restrict__ erel,
               const float* __restrict__ inv_from, const float* __restrict__ inv_to,
               float* __restrict__ acc, int E) {
  __shared__ __align__(16) float Rs[N_REL2 * D];
  const unsigned lbase = (unsigned)(uintptr_t)&Rs[0];
  for (int i = threadIdx.x; i < N_REL2 * D / 4; i += 256) {
    async_g2l_b128(lbase + (unsigned)(i * 16), RelT + i * 4);
  }
  async_wait0();
  __syncthreads();

  const int wv     = threadIdx.x >> 5;
  const int lane   = threadIdx.x & 31;
  const int stride = gridDim.x * 8;
  for (int e = blockIdx.x * 8 + wv; e < E; e += stride) {
    int e2 = e + stride;
    if (e2 < E) {                           // prefetch next edge's indices
      __builtin_prefetch(&efrom[e2], 0, 0);
      __builtin_prefetch(&eto[e2], 0, 0);
      __builtin_prefetch(&erel[e2], 0, 0);
    }
    int f = efrom[e], t = eto[e], r = erel[e];
    float s   = inv_from[f] * inv_to[t];
    float4 y  = Y[(size_t)f * 32 + lane];
    float4 rv = ((const float4*)Rs)[r * 32 + lane];
    float* d  = acc + (size_t)t * D + lane * 4;
    atomicAdd(d + 0, (y.x - rv.x) * s);
    atomicAdd(d + 1, (y.y - rv.y) * s);
    atomicAdd(d + 2, (y.z - rv.z) * s);
    atomicAdd(d + 3, (y.w - rv.w) * s);
  }
}

// --------------------------------------------- combine + (ReLU) + LayerNorm
// acc[n] = agg_out_scaled + agg_in_scaled + Y_loop[n].
// x = (acc - loopT)/3 + bias ; relu? ; LN(g,b).  1 wave per node row.
// Writes f16 (hidden layer) or f32 (final output).
__global__ __launch_bounds__(256)
void k_combine_ln(const float* __restrict__ acc, const float* __restrict__ loopT,
                  const float* __restrict__ bias, const float* __restrict__ lg,
                  const float* __restrict__ lb, _Float16* __restrict__ out_h,
                  float* __restrict__ out_f, int do_relu) {
  int row  = (blockIdx.x * 256 + threadIdx.x) >> 5;
  int lane = threadIdx.x & 31;
  if (row >= N_NODES) return;
  float4 v  = ((const float4*)(acc + (size_t)row * D))[lane];
  float4 lt = ((const float4*)loopT)[lane];
  float4 bi = ((const float4*)bias)[lane];
  const float c3 = 1.0f / 3.0f;
  v.x = (v.x - lt.x) * c3 + bi.x;
  v.y = (v.y - lt.y) * c3 + bi.y;
  v.z = (v.z - lt.z) * c3 + bi.z;
  v.w = (v.w - lt.w) * c3 + bi.w;
  if (do_relu) {
    v.x = fmaxf(v.x, 0.0f); v.y = fmaxf(v.y, 0.0f);
    v.z = fmaxf(v.z, 0.0f); v.w = fmaxf(v.w, 0.0f);
  }
  float s = v.x + v.y + v.z + v.w;
#pragma unroll
  for (int o = 16; o > 0; o >>= 1) s += __shfl_xor(s, o, 32);
  float mu = s * (1.0f / 128.0f);
  float dx = v.x - mu, dy = v.y - mu, dz = v.z - mu, dw = v.w - mu;
  float q = dx * dx + dy * dy + dz * dz + dw * dw;
#pragma unroll
  for (int o = 16; o > 0; o >>= 1) q += __shfl_xor(q, o, 32);
  float rstd = rsqrtf(q * (1.0f / 128.0f) + LN_EPS);
  float4 gg = ((const float4*)lg)[lane];
  float4 bb = ((const float4*)lb)[lane];
  float rx = dx * rstd * gg.x + bb.x;
  float ry = dy * rstd * gg.y + bb.y;
  float rz = dz * rstd * gg.z + bb.z;
  float rw = dw * rstd * gg.w + bb.w;
  if (out_f) {
    float4 o4 = {rx, ry, rz, rw};
    ((float4*)(out_f + (size_t)row * D))[lane] = o4;
  } else {
    v4h o4 = {(_Float16)rx, (_Float16)ry, (_Float16)rz, (_Float16)rw};
    *(v4h*)(out_h + (size_t)row * D + lane * 4) = o4;
  }
}

// ---------------------------------------------------------------- launcher
extern "C" void kernel_launch(void* const* d_in, const int* in_sizes, int n_in,
                              void* d_out, int out_size, void* d_ws, size_t ws_size,
                              hipStream_t stream) {
  const float* x_typeA    = (const float*)d_in[0];
  const float* W_mlp      = (const float*)d_in[1];
  const float* b_mlp      = (const float*)d_in[2];
  const float* emb_B      = (const float*)d_in[3];
  const float* rel_emb[2] = {(const float*)d_in[4],  (const float*)d_in[12]};
  const float* loop_rel[2]= {(const float*)d_in[5],  (const float*)d_in[13]};
  const float* w_loop[2]  = {(const float*)d_in[6],  (const float*)d_in[14]};
  const float* w_in[2]    = {(const float*)d_in[7],  (const float*)d_in[15]};
  const float* w_out[2]   = {(const float*)d_in[8],  (const float*)d_in[16]};
  const float* bias[2]    = {(const float*)d_in[9],  (const float*)d_in[17]};
  const float* ln_g[2]    = {(const float*)d_in[10], (const float*)d_in[18]};
  const float* ln_b[2]    = {(const float*)d_in[11], (const float*)d_in[19]};
  const int*   edge_src   = (const int*)d_in[20];
  const int*   edge_rel   = (const int*)d_in[21];
  const int*   edge_dst   = (const int*)d_in[22];
  float* out = (float*)d_out;

  char* p = (char*)d_ws;
  auto carve = [&p](size_t bytes) -> char* {
    char* q = p;
    p += (bytes + 255) & ~(size_t)255;
    return q;
  };
  float*    inv_s  = (float*)carve((size_t)N_NODES * 4);
  float*    inv_t  = (float*)carve((size_t)N_NODES * 4);
  _Float16* Wh_mlp = (_Float16*)carve((size_t)D * D_IN_A * 2);
  _Float16* Wh_out[2];  _Float16* Wh_in[2];  _Float16* Wh_loop[2];
  float* RelT_out[2];   float* RelT_in[2];   float* loopT[2];
  for (int l = 0; l < 2; ++l) {
    Wh_out[l]   = (_Float16*)carve((size_t)D * D * 2);
    Wh_in[l]    = (_Float16*)carve((size_t)D * D * 2);
    Wh_loop[l]  = (_Float16*)carve((size_t)D * D * 2);
    RelT_out[l] = (float*)carve((size_t)N_REL2 * D * 4);
    RelT_in[l]  = (float*)carve((size_t)N_REL2 * D * 4);
    loopT[l]    = (float*)carve((size_t)D * 4);
  }
  const size_t ND = (size_t)N_NODES * D;
  _Float16* XhA  = (_Float16*)carve((size_t)N_TYPE_A * D_IN_A * 2); // f16 typeA features
  _Float16* Xh0  = (_Float16*)carve(ND * 2);                        // f16 activations
  _Float16* Xh1  = (_Float16*)carve(ND * 2);
  float*    Yo   = (float*)carve(ND * 4);
  float*    Yi   = (float*)carve(ND * 4);
  float*    accb = (float*)carve(ND * 4);

  // ---- degrees & normalizations (edge structure is layer-invariant)
  k_zero_f32<<<CDIV(N_NODES, 256), 256, 0, stream>>>(inv_s, N_NODES);
  k_zero_f32<<<CDIV(N_NODES, 256), 256, 0, stream>>>(inv_t, N_NODES);
  k_degree<<<CDIV(N_EDGES, 256), 256, 0, stream>>>(edge_src, edge_dst, inv_s, inv_t, N_EDGES);
  k_invsqrt<<<CDIV(N_NODES, 256), 256, 0, stream>>>(inv_s, N_NODES);
  k_invsqrt<<<CDIV(N_NODES, 256), 256, 0, stream>>>(inv_t, N_NODES);

  // ---- f16 weight / activation tables
  k_f32_to_f16<<<CDIV(D * D_IN_A, 256), 256, 0, stream>>>(W_mlp, Wh_mlp, D * D_IN_A);
  k_f32_to_f16<<<CDIV(N_TYPE_A * D_IN_A, 256), 256, 0, stream>>>(x_typeA, XhA, N_TYPE_A * D_IN_A);
  // typeB embeddings go straight into the unified f16 node table
  k_f32_to_f16<<<CDIV(N_TYPE_B * D, 256), 256, 0, stream>>>(emb_B, Xh0 + (size_t)N_TYPE_A * D,
                                                            N_TYPE_B * D);
  for (int l = 0; l < 2; ++l) {
    k_f32_to_f16<<<CDIV(D * D, 256), 256, 0, stream>>>(w_out[l],  Wh_out[l],  D * D);
    k_f32_to_f16<<<CDIV(D * D, 256), 256, 0, stream>>>(w_in[l],   Wh_in[l],   D * D);
    k_f32_to_f16<<<CDIV(D * D, 256), 256, 0, stream>>>(w_loop[l], Wh_loop[l], D * D);
    // relation / loop transforms in full f32 (tiny)
    k_small_gemm<<<CDIV(N_REL2 * D, 128), 128, 0, stream>>>(rel_emb[l], w_out[l], RelT_out[l], N_REL2, D, D);
    k_small_gemm<<<CDIV(N_REL2 * D, 128), 128, 0, stream>>>(rel_emb[l], w_in[l],  RelT_in[l],  N_REL2, D, D);
    k_small_gemm<<<1, 128, 0, stream>>>(loop_rel[l], w_loop[l], loopT[l], 1, D, D);
  }

  // ---- typeA encoder -> rows [0, 50000) of Xh0 (f16 out)
  k_gemm_wmma<<<N_TYPE_A / 16, 256, 0, stream>>>(XhA, Wh_mlp, b_mlp, nullptr, Xh0, D_IN_A);

  // ---- two CompGCN layers
  const _Float16* Xin = Xh0;
  for (int l = 0; l < 2; ++l) {
    k_gemm_wmma<<<N_NODES / 16, 256, 0, stream>>>(Xin, Wh_out[l],  nullptr, Yo,   nullptr, D);
    k_gemm_wmma<<<N_NODES / 16, 256, 0, stream>>>(Xin, Wh_in[l],   nullptr, Yi,   nullptr, D);
    k_gemm_wmma<<<N_NODES / 16, 256, 0, stream>>>(Xin, Wh_loop[l], nullptr, accb, nullptr, D); // acc := Y_loop
    // out direction: acc[t] += (Yo[s] - RelT_out[r]) * inv_s[s] * inv_t[t]
    k_scatter<<<SCATTER_BLOCKS, 256, 0, stream>>>((const float4*)Yo, RelT_out[l],
                                                  edge_src, edge_dst, edge_rel,
                                                  inv_s, inv_t, accb, N_EDGES);
    // in direction:  acc[s] += (Yi[t] - RelT_in[r]) * inv_t[t] * inv_s[s]
    k_scatter<<<SCATTER_BLOCKS, 256, 0, stream>>>((const float4*)Yi, RelT_in[l],
                                                  edge_dst, edge_src, edge_rel,
                                                  inv_t, inv_s, accb, N_EDGES);
    k_combine_ln<<<CDIV(N_NODES, 8), 256, 0, stream>>>(accb, loopT[l], bias[l],
                                                       ln_g[l], ln_b[l],
                                                       (l == 0) ? Xh1 : nullptr,
                                                       (l == 0) ? nullptr : out,
                                                       (l == 0) ? 1 : 0);
    Xin = Xh1;
  }
}
